// GDNBlock_34720515621540
// MI455X (gfx1250) — compile-verified, matrix operand order checked by reading, not run on hardware
//
#include <hip/hip_runtime.h>
#include <math.h>

// ---------------- problem constants ----------------
#define B_   4
#define S_   2048
#define D_   1024
#define H_   16
#define DH_  64
#define M_   (B_ * S_)      // 8192 token rows
#define D4_  (4 * D_)       // 4096

typedef __bf16 v16bf __attribute__((ext_vector_type(16)));
typedef float  v8f   __attribute__((ext_vector_type(8)));
typedef unsigned int uv4 __attribute__((ext_vector_type(4)));

// low 32 bits of a generic pointer to LDS == byte offset in the wave's LDS
// allocation (aperture bits live in addr[63:32], ISA 10.2 aperture table)
__device__ inline unsigned lds_lo32(const void* p) { return (unsigned)(size_t)p; }

// two GLOBAL_LOAD_ASYNC_TO_LDS_B128 (GVS mode): mem = saddr + vaddr + ioffset,
// lds = vdst + ioffset. Tracked by ASYNCcnt; in-order completion.
__device__ inline void async_ld2(unsigned lds, unsigned goff, const __bf16* base) {
    asm volatile("global_load_async_to_lds_b128 %0, %1, %2\n\t"
                 "global_load_async_to_lds_b128 %0, %1, %2 offset:16"
                 :: "v"(lds), "v"(goff), "s"(base) : "memory");
}

// =====================================================================
// Weight convert + transpose:  W[K][N] f32  ->  Wt[N][K] bf16
// =====================================================================
__global__ __launch_bounds__(256) void wtrans_kernel(const float* __restrict__ W,
                                                     __bf16* __restrict__ Wt,
                                                     int K, int N) {
    long long i = (long long)blockIdx.x * 256 + threadIdx.x;   // over N*K, k fastest
    long long total = (long long)K * N;
    if (i < total) {
        int k = (int)(i % K);
        int n = (int)(i / K);
        Wt[i] = (__bf16)W[(size_t)k * N + n];
    }
}

// =====================================================================
// LayerNorm: one 256-thread block per row of 1024, bf16 out
// =====================================================================
__global__ __launch_bounds__(256) void ln_kernel(const float* __restrict__ x,
                                                 const float* __restrict__ g,
                                                 const float* __restrict__ b,
                                                 __bf16* __restrict__ out) {
    __shared__ float rs[256], rs2[256];
    int row = blockIdx.x, tid = threadIdx.x;
    float4 v = reinterpret_cast<const float4*>(x + (size_t)row * D_)[tid];
    rs[tid]  = v.x + v.y + v.z + v.w;
    rs2[tid] = v.x * v.x + v.y * v.y + v.z * v.z + v.w * v.w;
    __syncthreads();
    for (int off = 128; off > 0; off >>= 1) {
        if (tid < off) { rs[tid] += rs[tid + off]; rs2[tid] += rs2[tid + off]; }
        __syncthreads();
    }
    float mu   = rs[0] * (1.0f / D_);
    float var  = rs2[0] * (1.0f / D_) - mu * mu;
    float rstd = rsqrtf(var + 1e-5f);
    float4 gv = reinterpret_cast<const float4*>(g)[tid];
    float4 bv = reinterpret_cast<const float4*>(b)[tid];
    __bf16* o = out + (size_t)row * D_ + tid * 4;
    o[0] = (__bf16)((v.x - mu) * rstd * gv.x + bv.x);
    o[1] = (__bf16)((v.y - mu) * rstd * gv.y + bv.y);
    o[2] = (__bf16)((v.z - mu) * rstd * gv.z + bv.z);
    o[3] = (__bf16)((v.w - mu) * rstd * gv.w + bv.w);
}

// =====================================================================
// beta / gate projections: sigmoid(xn1 @ Wb), sigmoid(xn1 @ Wg); N=16
// =====================================================================
__global__ __launch_bounds__(64) void betagate_kernel(const __bf16* __restrict__ xn1,
                                                      const float* __restrict__ Wb,
                                                      const float* __restrict__ Wg,
                                                      float* __restrict__ beta,
                                                      float* __restrict__ gate) {
    __shared__ float xs[D_];
    int row = blockIdx.x, tid = threadIdx.x;
    for (int i = tid; i < D_; i += 64) xs[i] = (float)xn1[(size_t)row * D_ + i];
    __syncthreads();
    if (tid < 32) {
        const float* W = (tid < 16) ? Wb : Wg;
        int h = tid & 15;
        float acc = 0.f;
        for (int d = 0; d < D_; ++d) acc += xs[d] * W[d * H_ + h];
        float sg = 1.0f / (1.0f + expf(-acc));
        if (tid < 16) beta[(size_t)row * H_ + h] = sg;
        else          gate[(size_t)row * H_ + h] = sg;
    }
}

// =====================================================================
// L2-normalize k per (token, head) row of 64: one wave32 per row
// =====================================================================
__global__ __launch_bounds__(32) void knorm_kernel(float* __restrict__ k) {
    size_t base = (size_t)blockIdx.x * DH_;
    int lane = threadIdx.x;
    float e0 = k[base + 2 * lane];
    float e1 = k[base + 2 * lane + 1];
    float ss = e0 * e0 + e1 * e1;
    for (int off = 16; off > 0; off >>= 1) ss += __shfl_xor(ss, off, 32);
    float inv = 1.0f / (sqrtf(ss) + 1e-6f);
    k[base + 2 * lane]     = e0 * inv;
    k[base + 2 * lane + 1] = e1 * inv;
}

// =====================================================================
// Gated delta-rule scan, chunked+pipelined. grid = B*H blocks, 64 thr.
// Lane v owns state column S[:,v] in 64 VGPRs. Chunks of CH=8 steps:
// prefetch chunk c+1 into registers while computing chunk c out of
// double-buffered LDS; v/beta/gate stay in registers (per-lane/scalar).
// g_t is a per-head scalar so k^T(g*S) = g*(k^T S).
// =====================================================================
#define CH 8

__global__ __launch_bounds__(64) void scan_kernel(const float* __restrict__ q,
                                                  const float* __restrict__ k,
                                                  const float* __restrict__ v,
                                                  const float* __restrict__ beta,
                                                  const float* __restrict__ gate,
                                                  __bf16* __restrict__ out) {
    __shared__ float kb[2][CH][DH_];
    __shared__ float qb[2][CH][DH_];
    int bh = blockIdx.x;
    int b = bh >> 4, h = bh & 15;
    int tid = threadIdx.x;                     // v-column index
    float Sst[DH_];
#pragma unroll
    for (int j = 0; j < DH_; ++j) Sst[j] = 0.f;

    size_t rowbase = (size_t)b * S_ * D_ + (size_t)h * DH_ + tid;  // (b,t=0,h,tid)
    size_t bgbase  = (size_t)b * S_ * H_ + h;

    float kr[CH], qr[CH], vr[CH], br[CH], gr[CH];
    float vcur[CH], bcur[CH], gcur[CH];
#pragma unroll
    for (int s = 0; s < CH; ++s) {
        kr[s] = k[rowbase + (size_t)s * D_];
        qr[s] = q[rowbase + (size_t)s * D_];
        vr[s] = v[rowbase + (size_t)s * D_];
        br[s] = beta[bgbase + (size_t)s * H_];
        gr[s] = gate[bgbase + (size_t)s * H_];
    }
#pragma unroll
    for (int s = 0; s < CH; ++s) {
        kb[0][s][tid] = kr[s];
        qb[0][s][tid] = qr[s];
        vcur[s] = vr[s]; bcur[s] = br[s]; gcur[s] = gr[s];
    }
    __syncthreads();

    const int NC = S_ / CH;
    for (int c = 0; c < NC; ++c) {
        int p = c & 1;
        bool more = (c + 1 < NC);
        if (more) {                            // prefetch chunk c+1 (latency hidden)
            size_t off  = rowbase + (size_t)(c + 1) * CH * D_;
            size_t boff = bgbase  + (size_t)(c + 1) * CH * H_;
#pragma unroll
            for (int s = 0; s < CH; ++s) {
                kr[s] = k[off + (size_t)s * D_];
                qr[s] = q[off + (size_t)s * D_];
                vr[s] = v[off + (size_t)s * D_];
                br[s] = beta[boff + (size_t)s * H_];
                gr[s] = gate[boff + (size_t)s * H_];
            }
        }
#pragma unroll
        for (int s = 0; s < CH; ++s) {         // CH dependent recurrence steps
            float gt = gcur[s];
            float acc = 0.f;
#pragma unroll
            for (int j = 0; j < DH_; ++j) acc += kb[p][s][j] * Sst[j];
            float coef = bcur[s] * (vcur[s] - gt * acc);
            float o = 0.f;
#pragma unroll
            for (int j = 0; j < DH_; ++j) {
                float s2 = Sst[j] * gt + kb[p][s][j] * coef;
                Sst[j] = s2;
                o += qb[p][s][j] * s2;
            }
            out[rowbase + (size_t)(c * CH + s) * D_] = (__bf16)o;
        }
        __syncthreads();                       // everyone done reading buf p
        if (more) {
#pragma unroll
            for (int s = 0; s < CH; ++s) {
                kb[p ^ 1][s][tid] = kr[s];
                qb[p ^ 1][s][tid] = qr[s];
                vcur[s] = vr[s]; bcur[s] = br[s]; gcur[s] = gr[s];
            }
            __syncthreads();                   // next chunk visible
        }
    }
}

// =====================================================================
// bf16 WMMA GEMM: C[M,N] = act( A[M,K] * Bt[N,K]^T * scale + bias ) + res
// 128x128x32 tiles, 256 thr = 8 waves, wave -> 32x64 subtile (2x4 frags).
// Tiles move global->LDS via GLOBAL_LOAD_ASYNC_TO_LDS_B128 (ASYNCcnt),
// double-buffered: issue tile kt+1, s_wait_asynccnt 4 (in-order done)
// guarantees tile kt landed, barrier, ds_load frags, 8x WMMA.
// Fragment layout per CDNA5 ISA 7.12.2 (16-bit A/B): lane<16 holds
// K=k0..k0+7 and K=k0+16..k0+23 with k0 = (lane>=16)*8.
// =====================================================================
#define BM 128
#define BN 128
#define BK 32

union FragU { v16bf v; uv4 q[2]; };

__device__ inline v16bf ldfrag(const __bf16* rowp, int k0) {
    FragU f;
    f.q[0] = *reinterpret_cast<const uv4*>(rowp + k0);
    f.q[1] = *reinterpret_cast<const uv4*>(rowp + k0 + 16);
    return f.v;
}

__global__ __launch_bounds__(256) void gemm_bf16_kernel(const __bf16* __restrict__ A,
                                                        const __bf16* __restrict__ Bt,
                                                        float* __restrict__ Cf,
                                                        __bf16* __restrict__ Cb,
                                                        const float* __restrict__ bias,
                                                        const float* __restrict__ res,
                                                        int M, int N, int K,
                                                        float scale, int act) {
    __shared__ __bf16 As[2][BM * BK];
    __shared__ __bf16 Bs[2][BN * BK];
    int tid  = threadIdx.x;
    int lane = tid & 31, wave = tid >> 5;
    int wr = wave & 3;      // M quadrant (32 rows)
    int wc = wave >> 2;     // N half (64 cols)
    int bm = blockIdx.y * BM, bn = blockIdx.x * BN;

    v8f acc[2][4];
#pragma unroll
    for (int m = 0; m < 2; ++m)
#pragma unroll
        for (int n = 0; n < 4; ++n)
#pragma unroll
            for (int e = 0; e < 8; ++e) acc[m][n][e] = 0.f;

    int r  = tid >> 1;              // 0..127: tile row staged by this thread
    int cc = (tid & 1) << 4;        // 0 or 16: K-half staged (32B)
    unsigned aoff = (unsigned)(((size_t)(bm + r) * K + cc) * 2);   // bytes
    unsigned boff = (unsigned)(((size_t)(bn + r) * K + cc) * 2);
    unsigned lob  = (unsigned)((r * BK + cc) * 2);
    unsigned ldsA[2] = { lds_lo32(&As[0][0]) + lob, lds_lo32(&As[1][0]) + lob };
    unsigned ldsB[2] = { lds_lo32(&Bs[0][0]) + lob, lds_lo32(&Bs[1][0]) + lob };

    int mrow0 = wr * 32 + (lane & 15);
    int nrow0 = wc * 64 + (lane & 15);
    int k0    = (lane >> 4) << 3;   // 0 or 8

    // issue tile 0 (4 async b128 per thread)
    async_ld2(ldsA[0], aoff, A);
    async_ld2(ldsB[0], boff, Bt);

    int nk = K / BK;
    for (int kt = 0; kt < nk; ++kt) {
        int p = kt & 1;
        if (kt + 1 < nk) {
            aoff += BK * 2; boff += BK * 2;
            async_ld2(ldsA[p ^ 1], aoff, A);
            async_ld2(ldsB[p ^ 1], boff, Bt);
            asm volatile("s_wait_asynccnt 0x4" ::: "memory");  // older 4 (tile kt) done
        } else {
            asm volatile("s_wait_asynccnt 0x0" ::: "memory");
        }
        __syncthreads();            // all waves' tile-kt data visible

        const __bf16* Asp = &As[p][0];
        const __bf16* Bsp = &Bs[p][0];
        v16bf af[2], bfr[4];
        af[0] = ldfrag(Asp + mrow0 * BK, k0);
        af[1] = ldfrag(Asp + (mrow0 + 16) * BK, k0);
#pragma unroll
        for (int n = 0; n < 4; ++n) bfr[n] = ldfrag(Bsp + (nrow0 + 16 * n) * BK, k0);
#pragma unroll
        for (int m = 0; m < 2; ++m)
#pragma unroll
            for (int n = 0; n < 4; ++n)
                acc[m][n] = __builtin_amdgcn_wmma_f32_16x16x32_bf16(
                    false, af[m], false, bfr[n], (short)0, acc[m][n], false, false);
        __syncthreads();            // done reading buf p before its reuse
    }

    // epilogue: VGPR j holds M=j (lanes 0-15) / M=j+8 (lanes 16-31), N=lane%16
    int rsel = (lane >> 4) << 3;
#pragma unroll
    for (int m = 0; m < 2; ++m) {
#pragma unroll
        for (int n = 0; n < 4; ++n) {
            int col = bn + wc * 64 + n * 16 + (lane & 15);
            float bv = bias ? bias[col] : 0.f;
#pragma unroll
            for (int j = 0; j < 8; ++j) {
                int row = bm + wr * 32 + m * 16 + rsel + j;
                float val = acc[m][n][j] * scale + bv;
                if (act) val = 0.5f * val * (1.0f + erff(val * 0.70710678118654752f));
                size_t idx = (size_t)row * N + col;
                if (res) val += res[idx];
                if (Cf) Cf[idx] = val;
                if (Cb) Cb[idx] = (__bf16)val;
            }
        }
    }
}

// =====================================================================
// host launcher
// =====================================================================
extern "C" void kernel_launch(void* const* d_in, const int* in_sizes, int n_in,
                              void* d_out, int out_size, void* d_ws, size_t ws_size,
                              hipStream_t stream) {
    (void)in_sizes; (void)n_in; (void)out_size; (void)ws_size;
    const float* x     = (const float*)d_in[0];
    const float* Wq    = (const float*)d_in[1];
    const float* Wk    = (const float*)d_in[2];
    const float* Wv    = (const float*)d_in[3];
    const float* Wb    = (const float*)d_in[4];
    const float* Wg    = (const float*)d_in[5];
    const float* Wo    = (const float*)d_in[6];
    const float* bo    = (const float*)d_in[7];
    const float* ln1g  = (const float*)d_in[8];
    const float* ln1b  = (const float*)d_in[9];
    const float* ln2g  = (const float*)d_in[10];
    const float* ln2b  = (const float*)d_in[11];
    const float* w1    = (const float*)d_in[12];
    const float* b1    = (const float*)d_in[13];
    const float* w2    = (const float*)d_in[14];
    const float* b2    = (const float*)d_in[15];
    float* out = (float*)d_out;

    // ---- carve workspace (~265 MB) ----
    char* p = (char*)d_ws;
    auto alloc = [&](size_t bytes) -> void* {
        void* r = (void*)p;
        p += (bytes + 255) & ~(size_t)255;
        return r;
    };
    __bf16* wqT  = (__bf16*)alloc((size_t)D_ * D_ * 2);
    __bf16* wkT  = (__bf16*)alloc((size_t)D_ * D_ * 2);
    __bf16* wvT  = (__bf16*)alloc((size_t)D_ * D_ * 2);
    __bf16* woT  = (__bf16*)alloc((size_t)D_ * D_ * 2);
    __bf16* w1T  = (__bf16*)alloc((size_t)D_ * D4_ * 2);
    __bf16* w2T  = (__bf16*)alloc((size_t)D4_ * D_ * 2);
    __bf16* xn1  = (__bf16*)alloc((size_t)M_ * D_ * 2);
    float*  qf   = (float*) alloc((size_t)M_ * D_ * 4);
    float*  kf   = (float*) alloc((size_t)M_ * D_ * 4);
    float*  vf   = (float*) alloc((size_t)M_ * D_ * 4);
    float*  beta = (float*) alloc((size_t)M_ * H_ * 4);
    float*  gate = (float*) alloc((size_t)M_ * H_ * 4);
    __bf16* attn = (__bf16*)alloc((size_t)M_ * D_ * 2);
    float*  y1   = (float*) alloc((size_t)M_ * D_ * 4);
    __bf16* h2   = (__bf16*)alloc((size_t)M_ * D_ * 2);
    __bf16* ub   = (__bf16*)alloc((size_t)M_ * D4_ * 2);

    // ---- weight convert+transpose (bf16, N-major so B frags are K-runs) ----
    {
        dim3 blk(256);
        wtrans_kernel<<<dim3((D_ * D_) / 256), blk, 0, stream>>>(Wq, wqT, D_, D_);
        wtrans_kernel<<<dim3((D_ * D_) / 256), blk, 0, stream>>>(Wk, wkT, D_, D_);
        wtrans_kernel<<<dim3((D_ * D_) / 256), blk, 0, stream>>>(Wv, wvT, D_, D_);
        wtrans_kernel<<<dim3((D_ * D_) / 256), blk, 0, stream>>>(Wo, woT, D_, D_);
        wtrans_kernel<<<dim3((D_ * D4_) / 256), blk, 0, stream>>>(w1, w1T, D_, D4_);
        wtrans_kernel<<<dim3((D4_ * D_) / 256), blk, 0, stream>>>(w2, w2T, D4_, D_);
    }

    // ---- LN1 ----
    ln_kernel<<<dim3(M_), dim3(256), 0, stream>>>(x, ln1g, ln1b, xn1);

    // ---- Q/K/V projections (WMMA) ----
    dim3 gblk(256);
    dim3 g1024(D_ / BN, M_ / BM);
    gemm_bf16_kernel<<<g1024, gblk, 0, stream>>>(xn1, wqT, qf, nullptr, nullptr, nullptr,
                                                 M_, D_, D_, 0.125f /*Dh^-0.5*/, 0);
    gemm_bf16_kernel<<<g1024, gblk, 0, stream>>>(xn1, wkT, kf, nullptr, nullptr, nullptr,
                                                 M_, D_, D_, 1.0f, 0);
    gemm_bf16_kernel<<<g1024, gblk, 0, stream>>>(xn1, wvT, vf, nullptr, nullptr, nullptr,
                                                 M_, D_, D_, 1.0f, 0);

    // ---- beta / gate, k-normalize ----
    betagate_kernel<<<dim3(M_), dim3(64), 0, stream>>>(xn1, Wb, Wg, beta, gate);
    knorm_kernel<<<dim3(M_ * H_), dim3(32), 0, stream>>>(kf);

    // ---- sequential gated delta-rule scan (chunked/pipelined) ----
    scan_kernel<<<dim3(B_ * H_), dim3(64), 0, stream>>>(qf, kf, vf, beta, gate, attn);

    // ---- output projection + bo + residual(x) -> y1 ----
    gemm_bf16_kernel<<<g1024, gblk, 0, stream>>>(attn, woT, y1, nullptr, bo, x,
                                                 M_, D_, D_, 1.0f, 0);

    // ---- LN2 ----
    ln_kernel<<<dim3(M_), dim3(256), 0, stream>>>(y1, ln2g, ln2b, h2);

    // ---- MLP ----
    dim3 g4096(D4_ / BN, M_ / BM);
    gemm_bf16_kernel<<<g4096, gblk, 0, stream>>>(h2, w1T, nullptr, ub, b1, nullptr,
                                                 M_, D4_, D_, 1.0f, 1 /*exact gelu*/);
    gemm_bf16_kernel<<<g1024, gblk, 0, stream>>>(ub, w2T, out, nullptr, b2, y1,
                                                 M_, D_, D4_, 1.0f, 0);
}
